// TranspConvBlock_32392643347157
// MI455X (gfx1250) — matrix operand order repaired and checked by reading, not compiled, not run on hardware
//
#include <hip/hip_runtime.h>

// ---------------- problem constants ----------------
#define B_    4
#define CIN   256
#define COUT  256
#define H_    64
#define W_    64
#define HW    4096          // 64*64
#define MROWS 16384         // B*H*W
#define KDCN  2304          // CIN*9
#define KUP   1024          // CIN*4 taps (tap-major: k = tap*256 + ci)
#define HO    128
#define WO    128

typedef __attribute__((ext_vector_type(16))) __bf16         bf16x16;
typedef __attribute__((ext_vector_type(8)))  float          f32x8;
typedef __attribute__((ext_vector_type(8)))  unsigned short u16x8;
typedef __attribute__((ext_vector_type(16))) unsigned short u16x16;

__device__ __forceinline__ unsigned short f2bf_bits(float f) {
  unsigned u = __builtin_bit_cast(unsigned, f);
  u += 0x7FFFu + ((u >> 16) & 1u);          // round-to-nearest-even
  return (unsigned short)(u >> 16);
}

__device__ __forceinline__ bf16x16 mk_frag(u16x8 lo, u16x8 hi) {
  u16x16 t;
#pragma unroll
  for (int e = 0; e < 8; ++e) { t[e] = lo[e]; t[e + 8] = hi[e]; }
  return __builtin_bit_cast(bf16x16, t);
}

__device__ __forceinline__ bf16x16 ld_bfrag32(const unsigned short* p) {
  return __builtin_bit_cast(bf16x16, *(const u16x16*)p);   // 32B-aligned
}

// ---------------- K1: offset conv (27 ch, 3x3, pad1) ----------------
__global__ __launch_bounds__(256) void offset_conv_k(
    const float* __restrict__ x, const float* __restrict__ w_off,
    const float* __restrict__ b_off, float* __restrict__ om) {
  int t = blockIdx.x * blockDim.x + threadIdx.x;      // over B*27*HW
  if (t >= B_ * 27 * HW) return;
  int p = t & (HW - 1);
  int rest = t >> 12;
  int oc = rest % 27, b = rest / 27;
  int h = p >> 6, w = p & 63;
  float s = b_off[oc];
  const float* wo = w_off + (size_t)oc * CIN * 9;
  const float* xb = x + (size_t)b * CIN * HW;
  for (int ci = 0; ci < CIN; ++ci) {
    const float* xc = xb + (size_t)ci * HW;
    const float* wc = wo + (size_t)ci * 9;
#pragma unroll
    for (int kh = 0; kh < 3; ++kh) {
      int hh = h + kh - 1;
      if ((unsigned)hh >= (unsigned)H_) continue;
#pragma unroll
      for (int kw = 0; kw < 3; ++kw) {
        int ww = w + kw - 1;
        if ((unsigned)ww >= (unsigned)W_) continue;
        s += xc[hh * W_ + ww] * wc[kh * 3 + kw];
      }
    }
  }
  om[t] = s;
}

// ---------------- K2: deformable bilinear sampling -> bf16 im2col ----------
// cols[m][ci*9 + kap], m = ((b*64)+h)*64 + w
__global__ __launch_bounds__(256) void sample_cols_k(
    const float* __restrict__ x, const float* __restrict__ om,
    unsigned short* __restrict__ cols) {
  int t = blockIdx.x * blockDim.x + threadIdx.x;      // over MROWS*9
  if (t >= MROWS * 9) return;
  int kap = t % 9;
  int m = t / 9;
  int b = m >> 12, p = m & (HW - 1);
  int h = p >> 6, w = p & 63;
  const size_t omb = (size_t)b * 27 * HW;
  float dy = om[omb + (size_t)kap * HW + p];
  float dx = om[omb + (size_t)(9 + kap) * HW + p];
  float ms = om[omb + (size_t)(18 + kap) * HW + p];
  ms = 1.0f / (1.0f + __expf(-ms));                   // sigmoid mask
  int ki = kap / 3, kj = kap % 3;
  float pyf = dy + (float)h + (float)(ki - 1);
  float pxf = dx + (float)w + (float)(kj - 1);
  float y0f = floorf(pyf), x0f = floorf(pxf);
  float wy = pyf - y0f, wx = pxf - x0f;
  int y0 = (int)y0f, x0 = (int)x0f, y1 = y0 + 1, x1 = x0 + 1;
  bool vy0 = (y0 >= 0) && (y0 <= H_ - 1), vy1 = (y1 >= 0) && (y1 <= H_ - 1);
  bool vx0 = (x0 >= 0) && (x0 <= W_ - 1), vx1 = (x1 >= 0) && (x1 <= W_ - 1);
  int cy0 = min(max(y0, 0), H_ - 1), cy1 = min(max(y1, 0), H_ - 1);
  int cx0 = min(max(x0, 0), W_ - 1), cx1 = min(max(x1, 0), W_ - 1);
  float w00 = (1.f - wy) * (1.f - wx) * ((vy0 && vx0) ? ms : 0.f);
  float w01 = (1.f - wy) * wx * ((vy0 && vx1) ? ms : 0.f);
  float w10 = wy * (1.f - wx) * ((vy1 && vx0) ? ms : 0.f);
  float w11 = wy * wx * ((vy1 && vx1) ? ms : 0.f);
  int i00 = cy0 * W_ + cx0, i01 = cy0 * W_ + cx1;
  int i10 = cy1 * W_ + cx0, i11 = cy1 * W_ + cx1;
  const float* xb = x + (size_t)b * CIN * HW;
  unsigned short* crow = cols + (size_t)m * KDCN + kap;
  for (int ci = 0; ci < CIN; ++ci) {
    const float* xc = xb + (size_t)ci * HW;
    float v = w00 * xc[i00] + w01 * xc[i01] + w10 * xc[i10] + w11 * xc[i11];
    crow[(size_t)ci * 9] = f2bf_bits(v);
  }
}

// ---------------- K3: convert w_dcn to bf16 (natural [Co][Cin*9] = Bt[n][k])
__global__ __launch_bounds__(256) void cvt_wdcn_k(
    const float* __restrict__ w_dcn, unsigned short* __restrict__ wdb) {
  int t = blockIdx.x * blockDim.x + threadIdx.x;
  if (t >= COUT * KDCN) return;
  wdb[t] = f2bf_bits(w_dcn[t]);
}

// ---------------- K4: WMMA GEMM  out1[m][n] = cols * wdcn^T + bias ---------
__global__ __launch_bounds__(128) void gemm_dcn_k(
    const unsigned short* __restrict__ cols, const unsigned short* __restrict__ wdb,
    const float* __restrict__ bias, float* __restrict__ out1) {
  const int lane = threadIdx.x & 31;
  const int wave = threadIdx.x >> 5;
  const int wm = wave >> 1, wn = wave & 1;
  const int m0 = blockIdx.x * 64 + wm * 32;
  const int n0 = blockIdx.y * 64 + wn * 32;
  const int lhalf = lane >> 4, lrow = lane & 15;

  f32x8 acc[2][2];
#pragma unroll
  for (int i = 0; i < 2; ++i)
#pragma unroll
    for (int j = 0; j < 2; ++j) acc[i][j] = {};

  const unsigned short* aR0 = cols + (size_t)(m0 + lrow) * KDCN;
  const unsigned short* aR1 = cols + (size_t)(m0 + 16 + lrow) * KDCN;
  const unsigned short* bR0 = wdb + (size_t)(n0 + lrow) * KDCN;
  const unsigned short* bR1 = wdb + (size_t)(n0 + 16 + lrow) * KDCN;

#pragma unroll 2
  for (int kb = 0; kb < KDCN; kb += 32) {
    int ka = kb + lhalf * 8;
    bf16x16 a0 = mk_frag(*(const u16x8*)(aR0 + ka), *(const u16x8*)(aR0 + ka + 16));
    bf16x16 a1 = mk_frag(*(const u16x8*)(aR1 + ka), *(const u16x8*)(aR1 + ka + 16));
    int kbB = kb + lhalf * 8;
    bf16x16 b0 = mk_frag(*(const u16x8*)(bR0 + kbB), *(const u16x8*)(bR0 + kbB + 16));
    bf16x16 b1 = mk_frag(*(const u16x8*)(bR1 + kbB), *(const u16x8*)(bR1 + kbB + 16));
    acc[0][0] = __builtin_amdgcn_wmma_f32_16x16x32_bf16(false, a0, false, b0, (short)0, acc[0][0], false, false);
    acc[0][1] = __builtin_amdgcn_wmma_f32_16x16x32_bf16(false, a0, false, b1, (short)0, acc[0][1], false, false);
    acc[1][0] = __builtin_amdgcn_wmma_f32_16x16x32_bf16(false, a1, false, b0, (short)0, acc[1][0], false, false);
    acc[1][1] = __builtin_amdgcn_wmma_f32_16x16x32_bf16(false, a1, false, b1, (short)0, acc[1][1], false, false);
  }
#pragma unroll
  for (int i = 0; i < 2; ++i) {
    int mbase = m0 + i * 16 + lhalf * 8;
#pragma unroll
    for (int j = 0; j < 2; ++j) {
      int n = n0 + j * 16 + lrow;
      float bs = bias[n];
#pragma unroll
      for (int v = 0; v < 8; ++v) {
        int m = mbase + v;
        int b = m >> 12, p = m & (HW - 1);
        out1[((size_t)(b * COUT + n) << 12) + p] = acc[i][j][v] + bs;
      }
    }
  }
}

// ---------------- K5: per-channel BN statistics ----------------
__global__ __launch_bounds__(256) void bn_stats_k(
    const float* __restrict__ in, int C, int hw, int nb,
    float* __restrict__ mean, float* __restrict__ rstd) {
  __shared__ float s1[256], s2[256];
  int c = blockIdx.x;
  float a = 0.f, q = 0.f;
  for (int b = 0; b < nb; ++b) {
    const float* base = in + (size_t)(b * C + c) * hw;
    for (int p = threadIdx.x; p < hw; p += 256) {
      float v = base[p];
      a += v; q += v * v;
    }
  }
  s1[threadIdx.x] = a; s2[threadIdx.x] = q;
  __syncthreads();
  for (int o = 128; o > 0; o >>= 1) {
    if ((int)threadIdx.x < o) { s1[threadIdx.x] += s1[threadIdx.x + o]; s2[threadIdx.x] += s2[threadIdx.x + o]; }
    __syncthreads();
  }
  if (threadIdx.x == 0) {
    float R = (float)nb * (float)hw;
    float mn = s1[0] / R;
    float var = s2[0] / R - mn * mn;
    mean[c] = mn;
    rstd[c] = rsqrtf(var + 1e-5f);
  }
}

// ---------------- K6: BN1 + ReLU + convert to NHWC bf16 ----------------
__global__ __launch_bounds__(256) void bn1_act_cvt_k(
    const float* __restrict__ in, const float* __restrict__ mean,
    const float* __restrict__ rstd, const float* __restrict__ gamma,
    const float* __restrict__ beta, unsigned short* __restrict__ x1n) {
  int t = blockIdx.x * blockDim.x + threadIdx.x;      // B*C*HW
  if (t >= B_ * COUT * HW) return;
  int p = t & (HW - 1);
  int c = (t >> 12) & 255;
  int b = t >> 20;
  float y = (in[t] - mean[c]) * rstd[c] * gamma[c] + beta[c];
  y = fmaxf(y, 0.f);
  x1n[(((size_t)b * HW + p) << 8) + c] = f2bf_bits(y);
}

// ---------------- K7: rearrange w_up per parity class ----------------
// wub[pc][o][tap*256+i] = w_up[i][o][3-kh][3-kw], kh=2*th+py, kw=2*tw+px
__global__ __launch_bounds__(256) void prep_wup_k(
    const float* __restrict__ w_up, unsigned short* __restrict__ wub) {
  int t = blockIdx.x * blockDim.x + threadIdx.x;      // 4*256*1024
  if (t >= 4 * COUT * KUP) return;
  int pc = t >> 18;
  int o = (t >> 10) & 255;
  int k = t & 1023;
  int tap = k >> 8, ci = k & 255;
  int py = pc >> 1, px = pc & 1;
  int kh = 2 * (tap >> 1) + py;
  int kw = 2 * (tap & 1) + px;
  float v = w_up[(((size_t)ci * COUT + o) * 4 + (3 - kh)) * 4 + (3 - kw)];
  wub[t] = f2bf_bits(v);
}

// ---------------- K8: WMMA GEMM for transposed conv (per parity class) -----
__global__ __launch_bounds__(128) void gemm_up_k(
    const unsigned short* __restrict__ x1n, const unsigned short* __restrict__ wub,
    float* __restrict__ out) {
  const int pc = blockIdx.z;
  const int py = pc >> 1, px = pc & 1;
  const int lane = threadIdx.x & 31;
  const int wave = threadIdx.x >> 5;
  const int wm = wave >> 1, wn = wave & 1;
  const int m0 = blockIdx.x * 64 + wm * 32;
  const int n0 = blockIdx.y * 64 + wn * 32;
  const int lhalf = lane >> 4, lrow = lane & 15;
  const unsigned short* wB = wub + (size_t)pc * COUT * KUP;

  f32x8 acc[2][2];
#pragma unroll
  for (int i = 0; i < 2; ++i)
#pragma unroll
    for (int j = 0; j < 2; ++j) acc[i][j] = {};

  int nb[2], oh2[2], ow2[2];
#pragma unroll
  for (int r = 0; r < 2; ++r) {
    int m = m0 + r * 16 + lrow;
    nb[r] = m >> 12;
    int rem = m & (HW - 1);
    oh2[r] = rem >> 6; ow2[r] = rem & 63;
  }
  const unsigned short* bR0 = wB + (size_t)(n0 + lrow) * KUP;
  const unsigned short* bR1 = wB + (size_t)(n0 + 16 + lrow) * KUP;

#pragma unroll 2
  for (int kb = 0; kb < KUP; kb += 32) {
    bf16x16 afrag[2];
#pragma unroll
    for (int r = 0; r < 2; ++r) {
      u16x8 chunk[2];
#pragma unroll
      for (int c = 0; c < 2; ++c) {
        int k = kb + c * 16 + lhalf * 8;        // 8 contiguous k within one tap
        int tap = k >> 8, ib = k & 255;
        int h2 = oh2[r] + (tap >> 1) - 1 + py;
        int w2 = ow2[r] + (tap & 1) - 1 + px;
        u16x8 v;
        if ((unsigned)h2 < (unsigned)H_ && (unsigned)w2 < (unsigned)W_) {
          v = *(const u16x8*)(x1n + ((((size_t)nb[r] * H_ + h2) * W_ + w2) << 8) + ib);
        } else {
#pragma unroll
          for (int e = 0; e < 8; ++e) v[e] = 0;
        }
        chunk[c] = v;
      }
      afrag[r] = mk_frag(chunk[0], chunk[1]);
    }
    int kbB = kb + lhalf * 8;
    bf16x16 b0 = mk_frag(*(const u16x8*)(bR0 + kbB), *(const u16x8*)(bR0 + kbB + 16));
    bf16x16 b1 = mk_frag(*(const u16x8*)(bR1 + kbB), *(const u16x8*)(bR1 + kbB + 16));
    acc[0][0] = __builtin_amdgcn_wmma_f32_16x16x32_bf16(false, afrag[0], false, b0, (short)0, acc[0][0], false, false);
    acc[0][1] = __builtin_amdgcn_wmma_f32_16x16x32_bf16(false, afrag[0], false, b1, (short)0, acc[0][1], false, false);
    acc[1][0] = __builtin_amdgcn_wmma_f32_16x16x32_bf16(false, afrag[1], false, b0, (short)0, acc[1][0], false, false);
    acc[1][1] = __builtin_amdgcn_wmma_f32_16x16x32_bf16(false, afrag[1], false, b1, (short)0, acc[1][1], false, false);
  }
#pragma unroll
  for (int i = 0; i < 2; ++i) {
    int mbase = m0 + i * 16 + lhalf * 8;
#pragma unroll
    for (int j = 0; j < 2; ++j) {
      int n = n0 + j * 16 + lrow;
#pragma unroll
      for (int v = 0; v < 8; ++v) {
        int m = mbase + v;
        int b = m >> 12, rem = m & (HW - 1);
        int ho = ((rem >> 6) << 1) + py;
        int wo = ((rem & 63) << 1) + px;
        out[((size_t)(b * COUT + n) << 14) + ho * WO + wo] = acc[i][j][v];
      }
    }
  }
}

// ---------------- K9: BN2 + ReLU in place on d_out ----------------
__global__ __launch_bounds__(256) void bn2_act_k(
    float* __restrict__ out, const float* __restrict__ mean,
    const float* __restrict__ rstd, const float* __restrict__ gamma,
    const float* __restrict__ beta) {
  int t = blockIdx.x * blockDim.x + threadIdx.x;      // B*C*HO*WO = 2^24
  if (t >= B_ * COUT * HO * WO) return;
  int c = (t >> 14) & 255;
  float y = (out[t] - mean[c]) * rstd[c] * gamma[c] + beta[c];
  out[t] = fmaxf(y, 0.f);
}

// ---------------- host launcher ----------------
extern "C" void kernel_launch(void* const* d_in, const int* in_sizes, int n_in,
                              void* d_out, int out_size, void* d_ws, size_t ws_size,
                              hipStream_t stream) {
  (void)in_sizes; (void)n_in; (void)out_size; (void)ws_size;
  const float* x      = (const float*)d_in[0];
  const float* w_off  = (const float*)d_in[1];
  const float* b_off  = (const float*)d_in[2];
  const float* w_dcn  = (const float*)d_in[3];
  const float* b_dcn  = (const float*)d_in[4];
  const float* gamma1 = (const float*)d_in[5];
  const float* beta1  = (const float*)d_in[6];
  const float* w_up   = (const float*)d_in[7];
  const float* gamma2 = (const float*)d_in[8];
  const float* beta2  = (const float*)d_in[9];
  float* out = (float*)d_out;

  char* ws = (char*)d_ws;
  size_t off = 0;
  auto alloc = [&](size_t bytes) -> void* {
    void* p = ws + off;
    off = (off + bytes + 255) & ~(size_t)255;
    return p;
  };
  float*          om   = (float*)         alloc((size_t)B_ * 27 * HW * 4);
  unsigned short* cols = (unsigned short*)alloc((size_t)MROWS * KDCN * 2);
  unsigned short* wdb  = (unsigned short*)alloc((size_t)COUT * KDCN * 2);
  float*          out1 = (float*)         alloc((size_t)B_ * COUT * HW * 4);
  unsigned short* x1n  = (unsigned short*)alloc((size_t)B_ * HW * COUT * 2);
  unsigned short* wub  = (unsigned short*)alloc((size_t)4 * COUT * KUP * 2);
  float* mean1 = (float*)alloc(COUT * 4);
  float* rstd1 = (float*)alloc(COUT * 4);
  float* mean2 = (float*)alloc(COUT * 4);
  float* rstd2 = (float*)alloc(COUT * 4);

  // stage 1: offsets/mask
  offset_conv_k<<<dim3((B_ * 27 * HW + 255) / 256), dim3(256), 0, stream>>>(x, w_off, b_off, om);
  // stage 2: deformable bilinear sampling -> bf16 im2col
  sample_cols_k<<<dim3((MROWS * 9 + 255) / 256), dim3(256), 0, stream>>>(x, om, cols);
  cvt_wdcn_k<<<dim3((COUT * KDCN + 255) / 256), dim3(256), 0, stream>>>(w_dcn, wdb);
  // stage 3: DCN contraction via WMMA
  gemm_dcn_k<<<dim3(MROWS / 64, COUT / 64), dim3(128), 0, stream>>>(cols, wdb, b_dcn, out1);
  // stage 4: BN1 + ReLU -> NHWC bf16
  bn_stats_k<<<dim3(COUT), dim3(256), 0, stream>>>(out1, COUT, HW, B_, mean1, rstd1);
  bn1_act_cvt_k<<<dim3((B_ * COUT * HW + 255) / 256), dim3(256), 0, stream>>>(
      out1, mean1, rstd1, gamma1, beta1, x1n);
  // stage 5: transposed conv as 4 parity-class WMMA GEMMs
  prep_wup_k<<<dim3((4 * COUT * KUP + 255) / 256), dim3(256), 0, stream>>>(w_up, wub);
  gemm_up_k<<<dim3(MROWS / 64, COUT / 64, 4), dim3(128), 0, stream>>>(x1n, wub, out);
  // stage 6: BN2 + ReLU in place
  bn_stats_k<<<dim3(COUT), dim3(256), 0, stream>>>(out, COUT, HO * WO, B_, mean2, rstd2);
  bn2_act_k<<<dim3((B_ * COUT * HO * WO + 255) / 256), dim3(256), 0, stream>>>(
      out, mean2, rstd2, gamma2, beta2);
}
